// ScaledDotProductAttention_34849364639992
// MI455X (gfx1250) — compile-verified, matrix operand order checked by reading, not run on hardware
//
#include <hip/hip_runtime.h>

typedef __attribute__((ext_vector_type(16))) _Float16 v16h;
typedef __attribute__((ext_vector_type(8)))  float    v8f;
typedef __attribute__((ext_vector_type(4)))  float    v4f;

#define S_LEN   2048
#define D_DIM   64
#define QT      16      // q rows per workgroup
#define THREADS 256     // 8 wave32s

__device__ __forceinline__ float neg_inf() { return __int_as_float(0xff800000); }

// 16x32 f16 A-fragment (CDNA5 v16h layout) from row-major f32 matrix, leading dim ld.
// lane<16: row=lane,     K = k0+{0..7, 16..23}
// lane>=16: row=lane-16, K = k0+{8..15, 24..31}
__device__ __forceinline__ v16h load_a_f32(const float* __restrict__ base, int ld,
                                           int k0, int lane) {
  const int m    = lane & 15;
  const int koff = (lane >> 4) << 3;   // 0 or 8
  const float* p0 = base + m * ld + k0 + koff;
  const float* p1 = p0 + 16;
  v16h a;
#pragma unroll
  for (int j = 0; j < 8; ++j) a[j]     = (_Float16)p0[j];
#pragma unroll
  for (int j = 0; j < 8; ++j) a[8 + j] = (_Float16)p1[j];
  return a;
}

// 32x16 f16 B-fragment from row-major V: B[k][n] = V[k0+k][n0+n]
__device__ __forceinline__ v16h load_b_v(const float* __restrict__ vm,
                                         int k0, int n0, int lane) {
  const int n    = lane & 15;
  const int koff = (lane >> 4) << 3;
  const float* p0 = vm + (size_t)(k0 + koff) * D_DIM + n0 + n;
  const float* p1 = vm + (size_t)(k0 + 16 + koff) * D_DIM + n0 + n;
  v16h b;
#pragma unroll
  for (int j = 0; j < 8; ++j) b[j]     = (_Float16)p0[j * D_DIM];
#pragma unroll
  for (int j = 0; j < 8; ++j) b[8 + j] = (_Float16)p1[j * D_DIM];
  return b;
}

__global__ __launch_bounds__(THREADS, 1)
void attn_fused_kernel(const float* __restrict__ q, const float* __restrict__ k,
                       const float* __restrict__ v, const int* __restrict__ mask,
                       float* __restrict__ out, float* __restrict__ score) {
  __shared__ __align__(16) float sS[QT * S_LEN];  // 128 KB score strip -> probabilities
  __shared__ float sRed[16][16];
  __shared__ float sMax[16];
  __shared__ float sSum[16];
  __shared__ float sPart[4][32][8];    // phase-4 partial accumulators (waves 4..7)

  const int qt   = blockIdx.x;         // 0..127  (q tile)
  const int bh   = blockIdx.y;         // 0..31   (b*H + h)
  const int q0   = qt * QT;
  const int tid  = threadIdx.x;
  const int wave = tid >> 5;
  const int lane = tid & 31;
  const int n    = lane & 15;          // C-frag column
  const int mo   = (lane >> 4) << 3;   // C-frag row offset (0 or 8)

  const float* qBase = q + ((size_t)bh * S_LEN + q0) * D_DIM;
  const float* kBase = k + (size_t)bh * S_LEN * D_DIM;
  const float* vBase = v + (size_t)bh * S_LEN * D_DIM;

  // ---------- Phase 1: S = mask ? (Q K^T)/8 : -inf, staged to LDS ----------
  v16h aq0 = load_a_f32(qBase, D_DIM, 0,  lane);
  v16h aq1 = load_a_f32(qBase, D_DIM, 32, lane);

#pragma unroll 2
  for (int t = 0; t < 16; ++t) {
    const int kt   = wave * 16 + t;    // k-tile 0..127, disjoint per wave
    const int kcol = kt * 16 + n;
    const float* kTile = kBase + (size_t)kt * 16 * D_DIM;
    v16h b0 = load_a_f32(kTile, D_DIM, 0,  lane);  // B = K^T: same layout as A
    v16h b1 = load_a_f32(kTile, D_DIM, 32, lane);
    v8f c = {};
    c = __builtin_amdgcn_wmma_f32_16x16x32_f16(false, aq0, false, b0, (short)0, c, false, false);
    c = __builtin_amdgcn_wmma_f32_16x16x32_f16(false, aq1, false, b1, (short)0, c, false, false);
#pragma unroll
    for (int r = 0; r < 8; ++r) {
      const int row = mo + r;
      const float sc = (mask[(size_t)(q0 + row) * S_LEN + kcol] == 0)
                           ? neg_inf() : c[r] * 0.125f;
      sS[row * S_LEN + kt * 16 + n] = sc;
    }
  }
  __syncthreads();

  // ---------- Phase 2: row max / sum(exp), 16 threads per row, b128 LDS scans ----------
  const int row = tid & 15;
  const int seg = tid >> 4;            // 0..15, 128 cols each
  {
    const v4f* p = (const v4f*)&sS[row * S_LEN + seg * 128];
    float m = neg_inf();
#pragma unroll 4
    for (int i = 0; i < 32; ++i) {
      v4f x = p[i];
      m = fmaxf(m, fmaxf(fmaxf(x.x, x.y), fmaxf(x.z, x.w)));
    }
    sRed[row][seg] = m;
  }
  __syncthreads();
  if (tid < 16) {
    float m = sRed[tid][0];
    for (int i = 1; i < 16; ++i) m = fmaxf(m, sRed[tid][i]);
    sMax[tid] = m;
  }
  __syncthreads();
  {
    const float m = sMax[row];
    const v4f* p = (const v4f*)&sS[row * S_LEN + seg * 128];
    float s = 0.f;
#pragma unroll 4
    for (int i = 0; i < 32; ++i) {
      v4f x = p[i];
      s += __expf(x.x - m) + __expf(x.y - m) + __expf(x.z - m) + __expf(x.w - m);
    }
    sRed[row][seg] = s;
  }
  __syncthreads();
  if (tid < 16) {
    float s = 0.f;
    for (int i = 0; i < 16; ++i) s += sRed[tid][i];
    sSum[tid] = s;
  }
  __syncthreads();

  // ---------- Phase 3: normalize; stream P to HBM once, b128 non-temporal ----------
  // Score matrix is write-once/never-re-read: NT keeps the 536 MB stream from
  // evicting K/V/mask working sets out of the 192 MB L2.
  v4f* scoreV = (v4f*)(score + ((size_t)bh * S_LEN + q0) * S_LEN);
  v4f* sSV    = (v4f*)sS;
#pragma unroll 4
  for (int i4 = tid; i4 < QT * S_LEN / 4; i4 += THREADS) {
    const int r = i4 >> 9;             // row: 512 v4 per row (4*2048 bytes)
    const float m   = sMax[r];
    const float inv = 1.0f / sSum[r];
    v4f x = sSV[i4];
    v4f pv;
    pv.x = __expf(x.x - m) * inv;
    pv.y = __expf(x.y - m) * inv;
    pv.z = __expf(x.z - m) * inv;
    pv.w = __expf(x.w - m) * inv;
    sSV[i4] = pv;                      // keep P in LDS for phase 4
    __builtin_nontemporal_store(pv, &scoreV[i4]);
  }
  __syncthreads();

  // ---------- Phase 4: O = P @ V (4 n-tiles x 2 k-halves over 8 waves) ----------
  const int ntile = wave & 3;
  const int khalf = wave >> 2;
  v8f acc = {};
#pragma unroll 2
  for (int t = 0; t < 32; ++t) {
    const int k0 = khalf * 1024 + t * 32;
    v16h a = load_a_f32(sS, S_LEN, k0, lane);    // P from LDS
    v16h b = load_b_v(vBase, k0, ntile * 16, lane);
    acc = __builtin_amdgcn_wmma_f32_16x16x32_f16(false, a, false, b, (short)0, acc, false, false);
  }
  if (khalf == 1) {
#pragma unroll
    for (int r = 0; r < 8; ++r) sPart[ntile][lane][r] = acc[r];
  }
  __syncthreads();
  if (khalf == 0) {
    float* outBase = out + ((size_t)bh * S_LEN + q0) * D_DIM + ntile * 16;
#pragma unroll
    for (int r = 0; r < 8; ++r)
      __builtin_nontemporal_store(acc[r] + sPart[ntile][lane][r],
                                  &outBase[(size_t)(mo + r) * D_DIM + n]);
  }
}

extern "C" void kernel_launch(void* const* d_in, const int* in_sizes, int n_in,
                              void* d_out, int out_size, void* d_ws, size_t ws_size,
                              hipStream_t stream) {
  const float* q    = (const float*)d_in[0];
  const float* k    = (const float*)d_in[1];
  const float* v    = (const float*)d_in[2];
  const int*   mask = (const int*)d_in[3];
  float* out   = (float*)d_out;                         // [B,H,S,D] first
  float* score = out + (size_t)2 * 16 * S_LEN * D_DIM;  // then [B,H,S,S]
  dim3 grid(S_LEN / QT, 2 * 16);                        // 128 q-tiles x 32 (b,h)
  attn_fused_kernel<<<grid, THREADS, 0, stream>>>(q, k, v, mask, out, score);
}
